// Transform_Layer_53420803228147
// MI455X (gfx1250) — matrix operand order repaired and checked by reading, not compiled, not run on hardware
//
#include <hip/hip_runtime.h>

// FFT-based t-product: Y = ifft( fft(W) @ fft(x) ) + bias, all stages as WMMA GEMMs.
// L=128 tubal length, 65 unique real-FFT frequencies padded to KF=96 (zeros beyond 64).

typedef __attribute__((ext_vector_type(16))) _Float16 v16h;
typedef __attribute__((ext_vector_type(8)))  float    v8f;

#define L_TUBE 128
#define KF     96
#define MM     64
#define NN     64
#define PP     8192
#define NCOLS  (NN * PP)   // 524288

__device__ __forceinline__ v8f wmma_acc(v16h a, v16h b, v8f c) {
  // v_wmma_f32_16x16x32_f16 : D = A(16x32) * B(32x16) + C(16x16 f32)
  return __builtin_amdgcn_wmma_f32_16x16x32_f16(false, a, false, b, (short)0, c, false, false);
}

// A-fragment (16x32 f16, row-major source): lane holds row M = lane%16,
// half h = lane/16 selects K pattern: K(e) = h*16+e.
__device__ __forceinline__ v16h load_a(const _Float16* __restrict__ A, int lda,
                                       int row0, int col0, int lane) {
  const _Float16* p = A + (long)(row0 + (lane & 15)) * lda + col0;
  const int h = lane >> 4;
  v16h f;
#pragma unroll
  for (int e = 0; e < 16; ++e) {
    int k = h * 16 + e;
    f[e] = p[k];
  }
  return f;
}

// B-fragment from LDS stored transposed [col][K+pad]: lane n = lane%16, K = h*16+e,
// i.e. 16 contiguous f16 (32B aligned since ldb % 16 == 0 and k0 % 16 == 0).
__device__ __forceinline__ v16h load_b(const _Float16* Bt, int ldb,
                                       int col0, int k0, int lane) {
  return *(const v16h*)&Bt[(col0 + (lane & 15)) * ldb + k0 + (lane >> 4) * 16];
}

// ---------------------------------------------------------------------------
// Kernel 1: build DFT matrices (f16) + Wf = DFT(W) along l (fp32 accumulate).
// Fre/Fim: KFxL forward DFT (rows >64 zero).  Gre/Gim: L x KF inverse with
// Hermitian weights (1,2,...,2,1, then 0) / 128 folded in.  WfImN = -WfIm
// (FP16 WMMA has no A-negate: NEG={CNeg,0,0}), used for Yr = Wr*Xr + (-Wi)*Xi.
// ---------------------------------------------------------------------------
__global__ void k_setup(const float* __restrict__ W,
                        _Float16* Fre, _Float16* Fim,
                        _Float16* Gre, _Float16* Gim,
                        _Float16* WfRe, _Float16* WfIm, _Float16* WfImN) {
  const int t = blockIdx.x * 256 + threadIdx.x;
  const float STEP = 6.283185307179586f / 128.0f;

  if (t < KF * L_TUBE) {            // forward DFT matrix, exp(-2*pi*i*k*l/128)
    int k = t / L_TUBE, l = t % L_TUBE;
    float re = 0.f, im = 0.f;
    if (k <= 64) {
      float ang = STEP * (float)((k * l) & 127);   // exact period-128 reduction
      float s, c; __sincosf(ang, &s, &c);
      re = c; im = -s;
    }
    Fre[t] = (_Float16)re; Fim[t] = (_Float16)im;
  }

  if (t < L_TUBE * KF) {            // inverse DFT matrix with rfft weights
    int a = t / KF, k = t % KF;
    float w = (k == 0 || k == 64) ? 1.f : (k < 64 ? 2.f : 0.f);
    float ang = STEP * (float)((k * a) & 127);
    float s, c; __sincosf(ang, &s, &c);
    Gre[t] = (_Float16)(w * c * (1.0f / 128.0f));
    Gim[t] = (_Float16)(-w * s * (1.0f / 128.0f));
  }

  if (t < KF * MM * NN) {           // Wf[k,m,n] = sum_l exp(-i th k l) W[l,m,n]
    int k = t / (MM * NN), mn = t % (MM * NN);
    float re = 0.f, im = 0.f;
    if (k <= 64) {
      for (int l = 0; l < L_TUBE; ++l) {
        float ang = STEP * (float)((k * l) & 127);
        float s, c; __sincosf(ang, &s, &c);
        float wv = W[l * (MM * NN) + mn];
        re = fmaf(c, wv, re);
        im = fmaf(-s, wv, im);
      }
    }
    WfRe[t] = (_Float16)re; WfIm[t] = (_Float16)im; WfImN[t] = (_Float16)(-im);
  }
}

// ---------------------------------------------------------------------------
// Kernel 2: Xf(96 x 524288) = F(96x128) @ x(128 x 524288).  128-col tiles.
// ---------------------------------------------------------------------------
__global__ void k_fwd_dft(const float* __restrict__ x,
                          const _Float16* __restrict__ Fre,
                          const _Float16* __restrict__ Fim,
                          _Float16* __restrict__ XfRe,
                          _Float16* __restrict__ XfIm) {
  __shared__ __align__(32) _Float16 Bt[128 * 144];   // [col][K=128 +16 pad], 36.9 KB
  const int tid = threadIdx.x, lane = tid & 31, wave = tid >> 5;
  const long cbase = (long)blockIdx.x * 128;

  const int c = tid & 127;
  for (int r = tid >> 7; r < 128; r += 2)            // coalesced f32 load, f16 convert
    Bt[c * 144 + r] = (_Float16)x[(long)r * NCOLS + cbase + c];
  __syncthreads();

  const int n16 = lane & 15, h = lane >> 4;
  for (int idx = wave; idx < 48; idx += 8) {         // 6 m-tiles x 8 n-tiles
    const int mt = idx >> 3, nt = idx & 7;
    v8f ar = {}, ai = {};
#pragma unroll
    for (int ks = 0; ks < 4; ++ks) {
      v16h b  = load_b(Bt, 144, nt * 16, ks * 32, lane);
      v16h fr = load_a(Fre, L_TUBE, mt * 16, ks * 32, lane);
      v16h fi = load_a(Fim, L_TUBE, mt * 16, ks * 32, lane);
      ar = wmma_acc(fr, b, ar);
      ai = wmma_acc(fi, b, ai);
    }
    const long col = cbase + nt * 16 + n16;
#pragma unroll
    for (int r = 0; r < 8; ++r) {
      const int row = mt * 16 + h * 8 + r;
      XfRe[(long)row * NCOLS + col] = (_Float16)ar[r];
      XfIm[(long)row * NCOLS + col] = (_Float16)ai[r];
    }
  }
}

// ---------------------------------------------------------------------------
// Kernel 3: per-frequency complex matmul Yf[k](64xP) = Wf[k](64x64) @ Xf[k](64xP)
// ---------------------------------------------------------------------------
__global__ void k_freq_mm(const _Float16* __restrict__ WfRe,
                          const _Float16* __restrict__ WfIm,
                          const _Float16* __restrict__ WfImN,
                          const _Float16* __restrict__ XfRe,
                          const _Float16* __restrict__ XfIm,
                          _Float16* __restrict__ YfRe,
                          _Float16* __restrict__ YfIm) {
  __shared__ __align__(32) _Float16 BtR[128 * 80];   // [col][K=64 +16 pad]
  __shared__ __align__(32) _Float16 BtI[128 * 80];
  const int tid = threadIdx.x, lane = tid & 31, wave = tid >> 5;
  const int k = blockIdx.y;
  const long pbase = (long)blockIdx.x * 128;
  const long kofs  = (long)k * NCOLS;

  const int c = tid & 127;
  for (int n = tid >> 7; n < 64; n += 2) {
    BtR[c * 80 + n] = XfRe[kofs + (long)n * PP + pbase + c];
    BtI[c * 80 + n] = XfIm[kofs + (long)n * PP + pbase + c];
  }
  __syncthreads();

  const _Float16* Wr  = WfRe  + (long)k * (MM * NN);
  const _Float16* Wi  = WfIm  + (long)k * (MM * NN);
  const _Float16* Win = WfImN + (long)k * (MM * NN);
  const int n16 = lane & 15, h = lane >> 4;

  for (int idx = wave; idx < 32; idx += 8) {         // 4 m-tiles x 8 p-tiles
    const int mt = idx >> 3, pt = idx & 7;
    v8f yr = {}, yi = {};
#pragma unroll
    for (int ks = 0; ks < 2; ++ks) {
      v16h ar  = load_a(Wr,  NN, mt * 16, ks * 32, lane);
      v16h aip = load_a(Wi,  NN, mt * 16, ks * 32, lane);
      v16h ain = load_a(Win, NN, mt * 16, ks * 32, lane);
      v16h br  = load_b(BtR, 80, pt * 16, ks * 32, lane);
      v16h bi  = load_b(BtI, 80, pt * 16, ks * 32, lane);
      yr = wmma_acc(ar,  br, yr);
      yr = wmma_acc(ain, bi, yr);    // - Wi*Xi via negated operand
      yi = wmma_acc(ar,  bi, yi);
      yi = wmma_acc(aip, br, yi);
    }
    const long col = pbase + pt * 16 + n16;
#pragma unroll
    for (int r = 0; r < 8; ++r) {
      const int m = mt * 16 + h * 8 + r;
      YfRe[kofs + (long)m * PP + col] = (_Float16)yr[r];
      YfIm[kofs + (long)m * PP + col] = (_Float16)yi[r];
    }
  }
}

// ---------------------------------------------------------------------------
// Kernel 4: Y(128 x 524288) = Gre @ YfRe + Gim @ YfIm, + bias, f32 out.
// Column c = m*8192 + p matches both Yf's (96,64,8192) and Y's (128,64,8192).
// ---------------------------------------------------------------------------
__global__ void k_inv_dft(const _Float16* __restrict__ Gre,
                          const _Float16* __restrict__ Gim,
                          const _Float16* __restrict__ YfRe,
                          const _Float16* __restrict__ YfIm,
                          const float* __restrict__ bias,
                          float* __restrict__ Y) {
  __shared__ __align__(32) _Float16 BtR[128 * 112];  // [col][K=96 +16 pad]
  __shared__ __align__(32) _Float16 BtI[128 * 112];
  const int tid = threadIdx.x, lane = tid & 31, wave = tid >> 5;
  const long cbase = (long)blockIdx.x * 128;

  const int c = tid & 127;
  for (int kk = tid >> 7; kk < KF; kk += 2) {
    BtR[c * 112 + kk] = YfRe[(long)kk * NCOLS + cbase + c];
    BtI[c * 112 + kk] = YfIm[(long)kk * NCOLS + cbase + c];
  }
  __syncthreads();

  const int m = (int)(cbase >> 13);                  // 8192 % 128 == 0: one m per block
  const int n16 = lane & 15, h = lane >> 4;

  for (int idx = wave; idx < 64; idx += 8) {         // 8 a-tiles x 8 col-tiles
    const int at = idx >> 3, nt = idx & 7;
    v8f acc = {};
#pragma unroll
    for (int ks = 0; ks < 3; ++ks) {
      v16h gr = load_a(Gre, KF, at * 16, ks * 32, lane);
      v16h gi = load_a(Gim, KF, at * 16, ks * 32, lane);
      v16h br = load_b(BtR, 112, nt * 16, ks * 32, lane);
      v16h bi = load_b(BtI, 112, nt * 16, ks * 32, lane);
      acc = wmma_acc(gr, br, acc);
      acc = wmma_acc(gi, bi, acc);
    }
    const long col = cbase + nt * 16 + n16;
#pragma unroll
    for (int r = 0; r < 8; ++r) {
      const int a = at * 16 + h * 8 + r;
      Y[(long)a * NCOLS + col] = acc[r] + bias[a * MM + m];
    }
  }
}

// ---------------------------------------------------------------------------
extern "C" void kernel_launch(void* const* d_in, const int* in_sizes, int n_in,
                              void* d_out, int out_size, void* d_ws, size_t ws_size,
                              hipStream_t stream) {
  const float* x    = (const float*)d_in[0];   // (128, 64, 8192)
  const float* W    = (const float*)d_in[1];   // (128, 64, 64)
  const float* bias = (const float*)d_in[2];   // (128, 64, 1)
  float* Y = (float*)d_out;                    // (128, 64, 8192)

  char* p = (char*)d_ws;
  auto carve = [&](size_t elems) {
    _Float16* r = (_Float16*)p;
    p += ((elems * sizeof(_Float16) + 255) / 256) * 256;
    return r;
  };
  _Float16* Fre   = carve((size_t)KF * L_TUBE);
  _Float16* Fim   = carve((size_t)KF * L_TUBE);
  _Float16* Gre   = carve((size_t)L_TUBE * KF);
  _Float16* Gim   = carve((size_t)L_TUBE * KF);
  _Float16* WfRe  = carve((size_t)KF * MM * NN);
  _Float16* WfIm  = carve((size_t)KF * MM * NN);
  _Float16* WfImN = carve((size_t)KF * MM * NN);
  _Float16* XfRe  = carve((size_t)KF * NCOLS);
  _Float16* XfIm  = carve((size_t)KF * NCOLS);
  _Float16* YfRe  = carve((size_t)KF * NCOLS);
  _Float16* YfIm  = carve((size_t)KF * NCOLS);

  k_setup<<<(KF * MM * NN + 255) / 256, 256, 0, stream>>>(
      W, Fre, Fim, Gre, Gim, WfRe, WfIm, WfImN);
  k_fwd_dft<<<NCOLS / 128, 256, 0, stream>>>(x, Fre, Fim, XfRe, XfIm);
  k_freq_mm<<<dim3(PP / 128, KF), 256, 0, stream>>>(
      WfRe, WfIm, WfImN, XfRe, XfIm, YfRe, YfIm);
  k_inv_dft<<<NCOLS / 128, 256, 0, stream>>>(Gre, Gim, YfRe, YfIm, bias, Y);
}
